// KernelAxialMultiAttention_90778428769032
// MI455X (gfx1250) — matrix operand order repaired and checked by reading, not compiled, not run on hardware
//
#include <hip/hip_runtime.h>
#include <math.h>

// ---------------------------------------------------------------------------
// Problem constants (B=2, M=64, T=256, C=512, N=8 heads, D=64)
// ---------------------------------------------------------------------------
#define ROWS   32768          // B*M*T
#define COLS   512            // C
#define HEADS  8
#define HDIM   64
#define SEQT   256
#define NBM    128            // B*M

typedef __attribute__((ext_vector_type(16))) __bf16 v16bf;
typedef __attribute__((ext_vector_type(8)))  __bf16 v8bf;
typedef __attribute__((ext_vector_type(4)))  __bf16 v4bf;
typedef __attribute__((ext_vector_type(8)))  float  v8f;
typedef __attribute__((ext_vector_type(4)))  int    v4i;

#define TILE_M 128
#define TILE_N 128
#define TILE_K 32
#define LDS_STRIDE 48   // bf16 elems per staged row (96B => every 16B chunk aligned)

// ---------------------------------------------------------------------------
// CDNA5 async global->LDS copy (ASYNCcnt-tracked), with sync fallback
// ---------------------------------------------------------------------------
#if __has_builtin(__builtin_amdgcn_global_load_async_to_lds_b128) && \
    __has_builtin(__builtin_amdgcn_s_wait_asynccnt)
#define HAVE_ASYNC_LDS 1
#else
#define HAVE_ASYNC_LDS 0
#endif

typedef __attribute__((address_space(1))) v4i* gptr_v4i;
typedef __attribute__((address_space(3))) v4i* lptr_v4i;

__device__ __forceinline__ void async_copy_b128(const void* gsrc, void* ldst) {
#if HAVE_ASYNC_LDS
    __builtin_amdgcn_global_load_async_to_lds_b128(
        (gptr_v4i)gsrc, (lptr_v4i)ldst, 0, 0);
#else
    *(uint4*)ldst = *(const uint4*)gsrc;
#endif
}

__device__ __forceinline__ void async_wait_all() {
#if HAVE_ASYNC_LDS
    __builtin_amdgcn_s_wait_asynccnt(0);
#endif
}

// ---------------------------------------------------------------------------
// Kernel 0: f32 -> bf16 conversion (x and the three projection weights).
// Halves the GEMM's global read traffic and enables raw async-to-LDS staging.
// ---------------------------------------------------------------------------
__global__ __launch_bounds__(256)
void cvt_bf16_kernel(const float* __restrict__ src, __bf16* __restrict__ dst, int n4)
{
    const int i = blockIdx.x * 256 + threadIdx.x;
    if (i >= n4) return;
    float4 f = ((const float4*)src)[i];
    v4bf h;
    h[0] = (__bf16)f.x; h[1] = (__bf16)f.y; h[2] = (__bf16)f.z; h[3] = (__bf16)f.w;
    ((v4bf*)dst)[i] = h;
}

// ---------------------------------------------------------------------------
// Kernel 1: Y_z = act_z( x @ W_z^T + b_z )   for z in {q,k,v}
//   grid = (ROWS/128, COLS/128, 3), block = 256 (8 waves of 32)
//   bf16 inputs, double-buffered async-to-LDS staging,
//   wave tile = 32x64 = 2x4 v_wmma_f32_16x16x32_bf16 tiles
// ---------------------------------------------------------------------------
__global__ __launch_bounds__(256)
void qkv_gemm_kernel(const __bf16* __restrict__ xb,
                     const __bf16* __restrict__ wb_all,
                     const float* __restrict__ bq,
                     const float* __restrict__ bk,
                     const float* __restrict__ bv,
                     float* __restrict__ Yall)
{
    __shared__ __bf16 lds_a[2][TILE_M][LDS_STRIDE];
    __shared__ __bf16 lds_b[2][TILE_N][LDS_STRIDE];

    const int zi = blockIdx.z;
    const __bf16* __restrict__ wbz = wb_all + (size_t)zi * COLS * COLS;
    const float*  __restrict__ bias = (zi == 0) ? bq : ((zi == 1) ? bk : bv);
    float* __restrict__ Y = Yall + (size_t)zi * (size_t)ROWS * COLS;
    const bool act = (zi < 2);

    const int tid  = threadIdx.x;
    const int lane = tid & 31;
    const int wid  = tid >> 5;
    const int wm   = wid & 3;   // 4 waves along M -> 32 rows each
    const int wn   = wid >> 2;  // 2 waves along N -> 64 cols each
    const int rowBase = blockIdx.x * TILE_M;
    const int colBase = blockIdx.y * TILE_N;

    // staging: tile = 128 rows x 32 bf16 = 8 KB = 512 x 16B chunks.
    // 256 threads x 2 chunks; chunk c -> row = c>>2, koff = (c&3)*8 (bf16 elems)
    const int ch0_row = tid >> 2;
    const int ch0_ko  = (tid & 3) * 8;
    const int ch1_row = (tid + 256) >> 2;
    const int ch1_ko  = ((tid + 256) & 3) * 8;

    auto issue_tile = [&](int buf, int kk) {
        async_copy_b128(xb  + (size_t)(rowBase + ch0_row) * COLS + kk + ch0_ko,
                        &lds_a[buf][ch0_row][ch0_ko]);
        async_copy_b128(xb  + (size_t)(rowBase + ch1_row) * COLS + kk + ch1_ko,
                        &lds_a[buf][ch1_row][ch1_ko]);
        async_copy_b128(wbz + (size_t)(colBase + ch0_row) * COLS + kk + ch0_ko,
                        &lds_b[buf][ch0_row][ch0_ko]);
        async_copy_b128(wbz + (size_t)(colBase + ch1_row) * COLS + kk + ch1_ko,
                        &lds_b[buf][ch1_row][ch1_ko]);
    };

    v8f acc[2][4];
#pragma unroll
    for (int i = 0; i < 2; ++i)
#pragma unroll
        for (int j = 0; j < 4; ++j) {
            v8f zf = {0.f, 0.f, 0.f, 0.f, 0.f, 0.f, 0.f, 0.f};
            acc[i][j] = zf;
        }

    const int frow = lane & 15;   // fragment row/col within 16
    const int g    = lane >> 4;   // half-wave group

    issue_tile(0, 0);
    int cur = 0;

    for (int kk = 0; kk < COLS; kk += TILE_K) {
        async_wait_all();
        __syncthreads();
        if (kk + TILE_K < COLS) issue_tile(cur ^ 1, kk + TILE_K);

        // ---- A fragments: lane holds row (lane&15); K chunks g*8..+7, 16+g*8..+7 ----
        v16bf afrag[2];
#pragma unroll
        for (int i = 0; i < 2; ++i) {
            const __bf16* rp = &lds_a[cur][wm * 32 + i * 16 + frow][0];
            v8bf c0 = *(const v8bf*)(rp + g * 8);
            v8bf c1 = *(const v8bf*)(rp + 16 + g * 8);
            v16bf a;
#pragma unroll
            for (int e = 0; e < 8; ++e) { a[e] = c0[e]; a[8 + e] = c1[e]; }
            afrag[i] = a;
        }
        // ---- B fragments: lane holds col (lane&15); K chunks g*8..+7, 16+g*8..+7 ----
        v16bf bfrag[4];
#pragma unroll
        for (int j = 0; j < 4; ++j) {
            const __bf16* rp = &lds_b[cur][wn * 64 + j * 16 + frow][0];
            v8bf c0 = *(const v8bf*)(rp + g * 8);
            v8bf c1 = *(const v8bf*)(rp + 16 + g * 8);
            v16bf bvec;
#pragma unroll
            for (int e = 0; e < 8; ++e) { bvec[e] = c0[e]; bvec[8 + e] = c1[e]; }
            bfrag[j] = bvec;
        }

#pragma unroll
        for (int i = 0; i < 2; ++i)
#pragma unroll
            for (int j = 0; j < 4; ++j)
                acc[i][j] = __builtin_amdgcn_wmma_f32_16x16x32_bf16(
                    false, afrag[i], false, bfrag[j],
                    (short)0, acc[i][j], false, false);

        cur ^= 1;
    }

    // ---- epilogue: C/D layout — VGPR r holds M = r + 8*(lane>>4), N = lane&15 ----
    const int mhalf = (lane >> 4) * 8;
#pragma unroll
    for (int i = 0; i < 2; ++i) {
#pragma unroll
        for (int j = 0; j < 4; ++j) {
            const int col = colBase + wn * 64 + j * 16 + frow;
            const float bcol = bias[col];
#pragma unroll
            for (int r = 0; r < 8; ++r) {
                const int row = rowBase + wm * 32 + i * 16 + mhalf + r;
                float val = acc[i][j][r] + bcol;
                if (act) {
                    // branchless elu(y)+1 : y>0 ? y+1 : exp(y)
                    float ex = __expf(fminf(val, 0.f));
                    val = (val > 0.f) ? (val + 1.f) : ex;
                }
                Y[(size_t)row * COLS + col] = val;
            }
        }
    }
}

// ---------------------------------------------------------------------------
// Kernel 2: per-head stats.  grid = B*M*N = 1024, block = 256
//   ksum[d]    = sum_t k'[t,d]
//   rowsum[t]  = sum_d k'[t,d]
//   c[e]       = sum_t rowsum[t] * v[t,e]
//   z[t]       = 1 / (q'[t,:].ksum + eps)
// ---------------------------------------------------------------------------
__global__ __launch_bounds__(256)
void head_stats_kernel(const float* __restrict__ Yq,
                       const float* __restrict__ Yk,
                       const float* __restrict__ Yv,
                       float* __restrict__ z_out,
                       float* __restrict__ c_out)
{
    __shared__ float s_rowsum[SEQT];
    __shared__ float s_red[256];
    __shared__ float s_ksum[HDIM];

    const int h  = blockIdx.x;      // 0..1023 : (bm, n)
    const int bm = h >> 3;
    const int n  = h & 7;
    const size_t R = (size_t)bm * SEQT;
    const int Cb = n * HDIM;
    const int tid = threadIdx.x;

    // rowsum[t] : one thread per t, contiguous 256B read per thread
    {
        const float* kp = Yk + (R + tid) * COLS + Cb;
        float s = 0.f;
#pragma unroll
        for (int d = 0; d < HDIM; d += 4) {
            float4 f = *(const float4*)(kp + d);
            s += f.x + f.y + f.z + f.w;
        }
        s_rowsum[tid] = s;
    }

    // ksum partials: threads = (tg,d), coalesced over d
    const int d  = tid & 63;
    const int tg = tid >> 6;
    {
        float s = 0.f;
        for (int t = tg; t < SEQT; t += 4) s += Yk[(R + t) * COLS + Cb + d];
        s_red[tid] = s;
    }
    __syncthreads();
    if (tid < HDIM)
        s_ksum[tid] = s_red[tid] + s_red[tid + 64] + s_red[tid + 128] + s_red[tid + 192];
    __syncthreads();

    // c partials
    {
        float s = 0.f;
        for (int t = tg; t < SEQT; t += 4) s += s_rowsum[t] * Yv[(R + t) * COLS + Cb + d];
        __syncthreads();            // s_red consumers (ksum) done before overwrite
        s_red[tid] = s;
    }
    __syncthreads();
    if (tid < HDIM)
        c_out[(size_t)h * HDIM + tid] =
            s_red[tid] + s_red[tid + 64] + s_red[tid + 128] + s_red[tid + 192];

    // z[t]
    {
        const float* qp = Yq + (R + tid) * COLS + Cb;
        float dot = 0.f;
#pragma unroll 8
        for (int dd = 0; dd < HDIM; ++dd) dot += qp[dd] * s_ksum[dd];
        z_out[(size_t)h * SEQT + tid] = 1.0f / (dot + 1e-6f);
    }
}

// ---------------------------------------------------------------------------
// Kernel 3: rank-8 output projection per (b,m).  grid = 128, block = 256
//   W'[n][d'] = sum_e c[n][e] * Wp[d'][n*64+e]
//   out[t][d'] = bp[d'] + sum_n z[n][t] * W'[n][d']
// ---------------------------------------------------------------------------
__global__ __launch_bounds__(256)
void out_proj_kernel(const float* __restrict__ z_ws,
                     const float* __restrict__ c_ws,
                     const float* __restrict__ Wp,
                     const float* __restrict__ bp,
                     float* __restrict__ out)
{
    __shared__ float s_c[HEADS][HDIM];   // 2 KB
    __shared__ float s_wp[HEADS][COLS];  // 16 KB
    __shared__ float s_z[HEADS][SEQT];   // 8 KB

    const int bm  = blockIdx.x;
    const int tid = threadIdx.x;

    for (int i = tid; i < HEADS * HDIM; i += 256)
        s_c[i >> 6][i & 63] = c_ws[(size_t)bm * HEADS * HDIM + i];
    for (int i = tid; i < HEADS * SEQT; i += 256)
        s_z[i >> 8][i & 255] = z_ws[(size_t)bm * HEADS * SEQT + i];
    __syncthreads();

    for (int dp = tid; dp < COLS; dp += 256) {
        const float* wrow = Wp + (size_t)dp * COLS;
        __builtin_prefetch(wrow, 0, 3);       // global_prefetch_b8
#pragma unroll
        for (int n = 0; n < HEADS; ++n) {
            const float* wseg = wrow + n * HDIM;
            float s = 0.f;
#pragma unroll 8
            for (int e = 0; e < HDIM; ++e) s += s_c[n][e] * wseg[e];
            s_wp[n][dp] = s;
        }
    }
    __syncthreads();

    for (int dp = tid; dp < COLS; dp += 256) {
        float w0 = s_wp[0][dp], w1 = s_wp[1][dp], w2 = s_wp[2][dp], w3 = s_wp[3][dp];
        float w4 = s_wp[4][dp], w5 = s_wp[5][dp], w6 = s_wp[6][dp], w7 = s_wp[7][dp];
        const float bb = bp[dp];
        for (int t = 0; t < SEQT; ++t) {
            float v = bb
                    + s_z[0][t] * w0 + s_z[1][t] * w1
                    + s_z[2][t] * w2 + s_z[3][t] * w3
                    + s_z[4][t] * w4 + s_z[5][t] * w5
                    + s_z[6][t] * w6 + s_z[7][t] * w7;
            out[((size_t)bm * SEQT + t) * COLS + dp] = v;
        }
    }
}

// ---------------------------------------------------------------------------
// Host launcher
// ---------------------------------------------------------------------------
extern "C" void kernel_launch(void* const* d_in, const int* in_sizes, int n_in,
                              void* d_out, int out_size, void* d_ws, size_t ws_size,
                              hipStream_t stream) {
    const float* x  = (const float*)d_in[0];
    const float* Wq = (const float*)d_in[1];
    const float* bq = (const float*)d_in[2];
    const float* Wk = (const float*)d_in[3];
    const float* bk = (const float*)d_in[4];
    const float* Wv = (const float*)d_in[5];
    const float* bv = (const float*)d_in[6];
    const float* Wp = (const float*)d_in[7];
    const float* bp = (const float*)d_in[8];
    float* out = (float*)d_out;

    float* ws  = (float*)d_ws;
    const size_t planeSz = (size_t)ROWS * COLS;       // 16,777,216 floats
    float* Yq = ws;
    float* Yk = ws + planeSz;
    float* Yv = ws + 2 * planeSz;
    float* z_ws = ws + 3 * planeSz;                   // 1024*256 floats
    float* c_ws = z_ws + (size_t)NBM * HEADS * SEQT;  // 1024*64 floats
    float* bf_base = c_ws + (size_t)NBM * HEADS * HDIM;
    __bf16* xb16 = (__bf16*)bf_base;                  // ROWS*COLS bf16
    __bf16* wb16 = xb16 + planeSz;                    // 3 * COLS*COLS bf16

    // pass 0: convert inputs to bf16
    const int nx4 = (int)(planeSz / 4);               // 4,194,304
    const int nw4 = (COLS * COLS) / 4;                // 65,536
    cvt_bf16_kernel<<<dim3(nx4 / 256), 256, 0, stream>>>(x,  xb16, nx4);
    cvt_bf16_kernel<<<dim3(nw4 / 256), 256, 0, stream>>>(Wq, wb16,                 nw4);
    cvt_bf16_kernel<<<dim3(nw4 / 256), 256, 0, stream>>>(Wk, wb16 + COLS * COLS,     nw4);
    cvt_bf16_kernel<<<dim3(nw4 / 256), 256, 0, stream>>>(Wv, wb16 + 2 * COLS * COLS, nw4);

    // pass 1: WMMA GEMMs with async-to-LDS double buffering
    dim3 g1(ROWS / TILE_M, COLS / TILE_N, 3);
    qkv_gemm_kernel<<<g1, 256, 0, stream>>>(xb16, wb16, bq, bk, bv, ws);

    // pass 2: per-head linear-attention stats
    head_stats_kernel<<<dim3(NBM * HEADS), 256, 0, stream>>>(Yq, Yk, Yv, z_ws, c_ws);

    // pass 3: rank-8 output projection
    out_proj_kernel<<<dim3(NBM), 256, 0, stream>>>(z_ws, c_ws, Wp, bp, out);
}
